// O3EquivariantConv_56573309223684
// MI455X (gfx1250) — matrix operand order repaired and checked by reading, not compiled, not run on hardware
//
#include <hip/hip_runtime.h>
#include <hip/hip_bf16.h>
#include <math.h>

typedef __bf16 bf16;
typedef __attribute__((ext_vector_type(16))) __bf16 v16bf;
typedef __attribute__((ext_vector_type(8)))  __bf16 v8bf;
typedef __attribute__((ext_vector_type(8)))  float   v8f;

#define CIN   128
#define COUT  32
#define HIDN  32
#define NSH   18
#define KPAD  160   // 146 combined dims, padded to 5 chunks of 32 (with 2-slot gap at 18..19)
#define ROWSTR 168  // LDS row stride (bf16 elems), 336B: 16B aligned
#define FOFF  20    // features start at k=20 so their LDS stores are 8B-aligned
#define WAVES 8
#define MAXBLK 640  // persistent-wave grid cap (5120 waves)

// workspace layout (bf16 elements), fragments pre-swizzled per-lane:
//   wf1: [5 kc][2 nt][32 lane][16]  = 5120
//   wf2: [2 nt][32 lane][16]        = 1024
//   wf3: [2 nt][32 lane][16]        = 1024  (W3 folded over the 18 SH channels)
#define WF1_OFF 0
#define WF2_OFF 5120
#define WF3_OFF 6144
#define B3R_BYTE_OFF 14336  // float b3r[32] after 7168 bf16

__device__ __forceinline__ v8f wmma_bf16(v16bf a, v16bf b, v8f c) {
    return __builtin_amdgcn_wmma_f32_16x16x32_bf16(false, a, false, b, (short)0, c, false, false);
}

// fast SiLU: x * rcp(1+exp(-x)); v_exp_f32 + v_rcp_f32, no IEEE divide chain.
// Activations are immediately rounded to bf16, so approx rcp is ample precision.
__device__ __forceinline__ float silu_fast(float x) {
    return x * __builtin_amdgcn_rcpf(1.f + __expf(-x));
}

__global__ void zero_kernel(float* __restrict__ out, int n) {
    int i = blockIdx.x * blockDim.x + threadIdx.x;
    if (i < n) out[i] = 0.f;
}

// Fold W3 over SH channels, re-index W1 for the LDS combined layout, and
// swizzle all weights into per-lane WMMA B-fragment layout (bf16).
__global__ void preproc_kernel(const float* __restrict__ W1, const float* __restrict__ W2,
                               const float* __restrict__ W3, const float* __restrict__ b3,
                               bf16* __restrict__ wsb, float* __restrict__ b3r) {
    int tid = threadIdx.x;
    for (int t = tid; t < 5120; t += blockDim.x) {
        int j = t & 15, lane = (t >> 4) & 31, nt = (t >> 9) & 1, kc = t >> 10;
        int k = kc * 32 + (j >> 3) * 16 + (lane >> 4) * 8 + (j & 7);      // B frag: K = {g*8..g*8+7, 16+g*8..16+g*8+7} per lane
        int n = nt * 16 + (lane & 15);
        float v = 0.f;
        if (k < NSH)                          v = W1[k * HIDN + n];
        else if (k >= FOFF && k < FOFF + CIN) v = W1[(k - 2) * HIDN + n]; // feature f=k-20 -> W1 row 18+f
        wsb[WF1_OFF + t] = (bf16)v;
    }
    for (int t = tid; t < 1024; t += blockDim.x) {
        int j = t & 15, lane = (t >> 4) & 31, nt = t >> 9;
        int k = (j >> 3) * 16 + (lane >> 4) * 8 + (j & 7);
        int n = nt * 16 + (lane & 15);
        wsb[WF2_OFF + t] = (bf16)W2[k * HIDN + n];
        float acc = 0.f;
        for (int sh = 0; sh < NSH; ++sh) acc += W3[k * (NSH * COUT) + sh * COUT + n];
        wsb[WF3_OFF + t] = (bf16)acc;
    }
    for (int c = tid; c < COUT; c += blockDim.x) {
        float acc = 0.f;
        for (int sh = 0; sh < NSH; ++sh) acc += b3[sh * COUT + c];
        b3r[c] = acc;
    }
}

__global__ __launch_bounds__(256) void conv_kernel(
    const float* __restrict__ feat, const float* __restrict__ pos,
    const int* __restrict__ ei, const float* __restrict__ b1,
    const float* __restrict__ b2, const bf16* __restrict__ wsb,
    const float* __restrict__ b3r, float* __restrict__ out,
    int E, int nnodes, int ntiles, int totalWaves, int tilesPerWave)
{
    __shared__ bf16 sComb[WAVES][16][ROWSTR]; // combined rows, bf16, 16 edges/wave
    __shared__ bf16 sH1[WAVES][16][HIDN];
    __shared__ bf16 sH2[WAVES][16][HIDN];

    const int wave = threadIdx.x >> 5;
    const int lane = threadIdx.x & 31;
    const int g    = lane >> 4;     // half-wave group (WMMA K/M split)
    const int ml   = lane & 15;
    const int waveGid = blockIdx.x * WAVES + wave;

    // ---- weight fragments: loaded ONCE per wave, live in VGPRs across all tiles ----
    v16bf wf1[5][2], wf2[2], wf3[2];
    #pragma unroll
    for (int kc = 0; kc < 5; ++kc)
        #pragma unroll
        for (int nt = 0; nt < 2; ++nt)
            wf1[kc][nt] = *reinterpret_cast<const v16bf*>(&wsb[WF1_OFF + ((kc*2 + nt)*32 + lane)*16]);
    #pragma unroll
    for (int nt = 0; nt < 2; ++nt) {
        wf2[nt] = *reinterpret_cast<const v16bf*>(&wsb[WF2_OFF + (nt*32 + lane)*16]);
        wf3[nt] = *reinterpret_cast<const v16bf*>(&wsb[WF3_OFF + (nt*32 + lane)*16]);
    }
    const float b1v[2] = { b1[ml], b1[ml + 16] };
    const float b2v[2] = { b2[ml], b2[ml + 16] };
    const float b3v[2] = { b3r[ml], b3r[ml + 16] };

    for (int t = 0; t < tilesPerWave; ++t) {
        const int tile = waveGid + t * totalWaves;   // strided: adjacent waves on adjacent tiles
        const bool active = tile < ntiles;

        // ---- stage SH + indices (lanes 0..15: one edge each); intra-wave LDS only ----
        int colIdx = 0, rowIdx = -1;
        if (active && lane < 16) {
            int e = tile * 16 + lane;
            bool valid = e < E;
            if (!valid) e = E - 1;
            int r = ei[e];
            int c = ei[E + e];
            colIdx = c;
            rowIdx = valid ? r : -1;
            float px = pos[r*3+0] - pos[c*3+0];
            float py = pos[r*3+1] - pos[c*3+1];
            float pz = pos[r*3+2] - pos[c*3+2];
            float d = fmaxf(sqrtf(px*px + py*py + pz*pz), 1e-8f);
            float inv = __builtin_amdgcn_rcpf(d);
            float x = px * inv, y = py * inv, z = pz * inv;
            const float c0 = 0.28209479177387814f, c1a = 0.34549414947133547f,
                        c1b = 0.4886025119029199f, c2a = 0.3862742020231896f,
                        c2b = 0.7725484040463791f, c2c = 0.31539156525252005f;
            float s0 = c0, s1 = -c1a * x, s2 = c1b * z, s3 = c1a * x;
            float s4 = c2a * (x*x - y*y), s5 = -c2b * z * x;
            float s6 = c2c * (2.f*z*z - x*x - y*y), s7 = c2b * z * x, s8 = s4;
            bf16* rp = &sComb[wave][lane][0];
            rp[0]=(bf16)s0;  rp[1]=(bf16)s1;  rp[2]=(bf16)s2;  rp[3]=(bf16)s3;  rp[4]=(bf16)s4;
            rp[5]=(bf16)s5;  rp[6]=(bf16)s6;  rp[7]=(bf16)s7;  rp[8]=(bf16)s8;
            // SH(-u): l=1 terms flip sign, l=0/2 even
            rp[9]=(bf16)s0;  rp[10]=(bf16)(-s1); rp[11]=(bf16)(-s2); rp[12]=(bf16)(-s3);
            rp[13]=(bf16)s4; rp[14]=(bf16)s5;   rp[15]=(bf16)s6;    rp[16]=(bf16)s7; rp[17]=(bf16)s8;
            rp[18]=(bf16)0.f; rp[19]=(bf16)0.f;
            #pragma unroll
            for (int k = FOFF + CIN; k < KPAD; ++k) rp[k] = (bf16)0.f;  // K padding
        }
        // ---- gather features[col] (coalesced float4 per lane), convert to bf16 ----
        if (active) {
            #pragma unroll 4
            for (int m = 0; m < 16; ++m) {
                int cidx = __shfl(colIdx, m, 32);
                float4 f = *reinterpret_cast<const float4*>(&feat[(long)cidx * CIN + lane * 4]);
                struct __align__(8) BF4 { bf16 a, b, c, d; };
                BF4 p; p.a = (bf16)f.x; p.b = (bf16)f.y; p.c = (bf16)f.z; p.d = (bf16)f.w;
                *reinterpret_cast<BF4*>(&sComb[wave][m][FOFF + lane * 4]) = p; // 8B-aligned
            }
        }
        // intra-wave: DS pipe is in-order; wait + compiler barrier before fragment reads
        asm volatile("s_wait_dscnt 0x0" ::: "memory");

        // ---- layer 1: [16 x 160] @ [160 x 32]; bias folded into C-init; 10 WMMAs ----
        v8f acc0, acc1;
        #pragma unroll
        for (int r = 0; r < 8; ++r) { acc0[r] = b1v[0]; acc1[r] = b1v[1]; }
        #pragma unroll
        for (int kc = 0; kc < 5; ++kc) {
            // A frag 16-bit layout: lane half g holds K = {g*8..g*8+7, 16+g*8..16+g*8+7}
            v8bf alo = *reinterpret_cast<const v8bf*>(&sComb[wave][ml][kc*32 + g*8]);
            v8bf ahi = *reinterpret_cast<const v8bf*>(&sComb[wave][ml][kc*32 + 16 + g*8]);
            v16bf a = __builtin_shufflevector(alo, ahi, 0,1,2,3,4,5,6,7,8,9,10,11,12,13,14,15);
            acc0 = wmma_bf16(a, wf1[kc][0], acc0);
            acc1 = wmma_bf16(a, wf1[kc][1], acc1);
        }
        #pragma unroll
        for (int r = 0; r < 8; ++r) {   // D layout: lane holds (M = r + 8*g, N = ml + 16*nt)
            sH1[wave][r + 8*g][ml]      = (bf16)silu_fast(acc0[r]);
            sH1[wave][r + 8*g][ml + 16] = (bf16)silu_fast(acc1[r]);
        }
        asm volatile("s_wait_dscnt 0x0" ::: "memory");

        // ---- layer 2: [16 x 32] @ [32 x 32] ----
        #pragma unroll
        for (int r = 0; r < 8; ++r) { acc0[r] = b2v[0]; acc1[r] = b2v[1]; }
        {
            v8bf alo = *reinterpret_cast<const v8bf*>(&sH1[wave][ml][g*8]);
            v8bf ahi = *reinterpret_cast<const v8bf*>(&sH1[wave][ml][16 + g*8]);
            v16bf a = __builtin_shufflevector(alo, ahi, 0,1,2,3,4,5,6,7,8,9,10,11,12,13,14,15);
            acc0 = wmma_bf16(a, wf2[0], acc0);
            acc1 = wmma_bf16(a, wf2[1], acc1);
        }
        #pragma unroll
        for (int r = 0; r < 8; ++r) {
            sH2[wave][r + 8*g][ml]      = (bf16)silu_fast(acc0[r]);
            sH2[wave][r + 8*g][ml + 16] = (bf16)silu_fast(acc1[r]);
        }
        asm volatile("s_wait_dscnt 0x0" ::: "memory");

        // ---- layer 3 (SH-folded): [16 x 32] @ [32 x 32], then scatter-add ----
        #pragma unroll
        for (int r = 0; r < 8; ++r) { acc0[r] = b3v[0]; acc1[r] = b3v[1]; }
        {
            v8bf alo = *reinterpret_cast<const v8bf*>(&sH2[wave][ml][g*8]);
            v8bf ahi = *reinterpret_cast<const v8bf*>(&sH2[wave][ml][16 + g*8]);
            v16bf a = __builtin_shufflevector(alo, ahi, 0,1,2,3,4,5,6,7,8,9,10,11,12,13,14,15);
            acc0 = wmma_bf16(a, wf3[0], acc0);
            acc1 = wmma_bf16(a, wf3[1], acc1);
        }
        if (active) {
            #pragma unroll
            for (int r = 0; r < 8; ++r) {
                int rowm = __shfl(rowIdx, r + 8*g, 32);   // row index held by lane (r+8g)
                if ((unsigned)rowm < (unsigned)nnodes) {
                    atomicAdd(&out[(long)rowm * COUT + ml],      acc0[r]);
                    atomicAdd(&out[(long)rowm * COUT + ml + 16], acc1[r]);
                }
            }
        }
    }
}

extern "C" void kernel_launch(void* const* d_in, const int* in_sizes, int n_in,
                              void* d_out, int out_size, void* d_ws, size_t ws_size,
                              hipStream_t stream) {
    (void)n_in; (void)out_size; (void)ws_size;
    const float* feat = (const float*)d_in[0];
    const float* pos  = (const float*)d_in[1];
    const int*   ei   = (const int*)d_in[2];
    const float* W1   = (const float*)d_in[3];
    const float* b1   = (const float*)d_in[4];
    const float* W2   = (const float*)d_in[5];
    const float* b2   = (const float*)d_in[6];
    const float* W3   = (const float*)d_in[7];
    const float* b3   = (const float*)d_in[8];
    float* out = (float*)d_out;

    int E      = in_sizes[2] / 2;
    int nnodes = in_sizes[0] / CIN;
    bf16*  wsb = (bf16*)d_ws;
    float* b3r = (float*)((char*)d_ws + B3R_BYTE_OFF);

    int nOut = nnodes * COUT;
    zero_kernel<<<(nOut + 255) / 256, 256, 0, stream>>>(out, nOut);
    preproc_kernel<<<1, 256, 0, stream>>>(W1, W2, W3, b3, wsb, b3r);

    int ntiles = (E + 15) / 16;
    int nblk   = (ntiles + WAVES - 1) / WAVES;
    if (nblk > MAXBLK) nblk = MAXBLK;
    int totalWaves   = nblk * WAVES;
    int tilesPerWave = (ntiles + totalWaves - 1) / totalWaves;
    conv_kernel<<<nblk, 256, 0, stream>>>(feat, pos, ei, b1, b2, wsb, b3r, out,
                                          E, nnodes, ntiles, totalWaves, tilesPerWave);
}
